// CostVolume_17738214932883
// MI455X (gfx1250) — matrix operand order repaired and checked by reading, not compiled, not run on hardware
//
#include <hip/hip_runtime.h>
#include <hip/hip_bf16.h>

// ---------------------------------------------------------------------------
// Windowed correlation cost volume on gfx1250 via v_wmma_f32_16x16x32_f16.
//   f1, f2 : [B=8, C=256, H=96, W=160] fp32
//   out    : [B, 81, H, W] fp32,  out[b,(dy+4)*9+(dx+4),h,w] =
//            sum_c f1[b,c,h,w] * f2[b,c,(h-dy)%H,(w-dx)%W]
// Mapping: per (b, 4 h-rows, w-tile of 16) stage fp32 -> fp16 through LDS
// (w-major global -> channel-major fragments), then each of 9 waves (one per
// dy) computes the |w-w'|<=4 Gram band as two 16x16 WMMA tiles per h-row.
// Staging is channel-grouped: 8 coalesced b32 loads -> one ds_store_b128.
// ---------------------------------------------------------------------------

#define CV_B   8
#define CV_C   256
#define CV_H   96
#define CV_W   160
#define CV_HW  (CV_H * CV_W)
#define CV_K81 81
#define HPB    4            // output h-rows per block
#define NWAVE  9            // one wave per dy index
#define NTHR   (NWAVE * 32) // 288
#define LROW   264          // halfs per LDS row: 256 + 8 pad (528 B)

#define A_ROWS (HPB * 16)       // 64  rows: [h'][m]
#define B_ROWS ((HPB + 8) * 32) // 384 rows: [hr 0..11][j 0..31]
#define A_HALF (A_ROWS * LROW)  // 16896 halfs
#define B_HALF (B_ROWS * LROW)  // 101376 halfs
#define SMEM_BYTES ((A_HALF + B_HALF) * 2) // 236544 B (< 320 KB/WGP)

typedef _Float16 v16h  __attribute__((ext_vector_type(16)));
typedef _Float16 half8 __attribute__((ext_vector_type(8)));
typedef float    v8f   __attribute__((ext_vector_type(8)));

union Frag16 { v16h v; half8 p[2]; };

__global__ __launch_bounds__(NTHR) void
cost_volume_wmma_kernel(const float* __restrict__ f1,
                        const float* __restrict__ f2,
                        float* __restrict__ out) {
  extern __shared__ _Float16 smem[];
  _Float16* sA = smem;           // [64][LROW]  f1 tile, fp16
  _Float16* sB = smem + A_HALF;  // [384][LROW] f2 halo rows, fp16

  const int tid = threadIdx.x;
  const int w0  = blockIdx.x * 16;   // 10 tiles
  const int h0  = blockIdx.y * HPB;  // 24 blocks
  const int b   = blockIdx.z;        // 8

  // ---- Stage f1: 2048 groups of (h', m, 8 channels). Lanes vary m -> each
  // of the 8 strided b32 loads is coalesced; one ds_store_b128 per group.
  for (int g = tid; g < A_ROWS * (CV_C / 8); g += NTHR) {
    const int m  = g & 15;
    const int cg = (g >> 4) & 31;          // channel group: c = 8*cg .. 8*cg+7
    const int hq = g >> 9;
    const float* src = f1 + ((b * CV_C + cg * 8) * CV_H + (h0 + hq)) * CV_W
                          + (w0 + m);
    half8 hv;
#pragma unroll
    for (int q = 0; q < 8; ++q) hv[q] = (_Float16)src[q * CV_HW];
    *(half8*)(sA + (hq * 16 + m) * LROW + cg * 8) = hv;
  }

  // ---- Stage f2 halo: rows h0-4..h0+7, cols w0-4..w0+27 (mod).
  // 12288 groups of (hr, j, 8 channels); lanes vary j -> coalesced loads.
  for (int g = tid; g < B_ROWS * (CV_C / 8); g += NTHR) {
    const int j  = g & 31;
    const int cg = (g >> 5) & 31;
    const int hr = g >> 10;                 // 0..11
    int ww = w0 - 4 + j;  if (ww < 0) ww += CV_W; else if (ww >= CV_W) ww -= CV_W;
    int hh = h0 - 4 + hr; if (hh < 0) hh += CV_H; else if (hh >= CV_H) hh -= CV_H;
    const float* src = f2 + ((b * CV_C + cg * 8) * CV_H + hh) * CV_W + ww;
    half8 hv;
#pragma unroll
    for (int q = 0; q < 8; ++q) hv[q] = (_Float16)src[q * CV_HW];
    *(half8*)(sB + (hr * 32 + j) * LROW + cg * 8) = hv;
  }

  __syncthreads();

  // ---- Compute: wave wv handles dy-index wv (shift dy = wv-4).
  const int lane = tid & 31;
  const int wv   = tid >> 5;       // 0..8
  const int n    = lane & 15;      // fragment column / A row (per ISA layout)
  const int hi   = lane >> 4;      // lane half select

  for (int hq = 0; hq < HPB; ++hq) {
    // f2 row needed: h - (wv-4) = h0 + hq + 4 - wv  ->  staged index hq+8-wv.
    const int hr = hq + 8 - wv;    // 0..11
    v8f acc0 = {};                 // column tile at offset -4  (j = n)
    v8f acc1 = {};                 // column tile at offset +12 (j = 16+n)

    const _Float16* aBase  = sA + (hq * 16 + n) * LROW + 16 * hi;
    const _Float16* b0Base = sB + (hr * 32 + n) * LROW + 16 * hi;
    const _Float16* b1Base = sB + (hr * 32 + 16 + n) * LROW + 16 * hi;

#pragma unroll
    for (int kk = 0; kk < CV_C / 32; ++kk) {
      const int kb = kk * 32;
      Frag16 a, b0, b1;
      // A 16x32 f16: lane holds 16 consecutive K at 16*hi for row n.
      a.p[0] = *(const half8*)(aBase + kb);
      a.p[1] = *(const half8*)(aBase + kb + 8);
      // B 32x16 f16: lane holds 16 consecutive K at 16*hi for column n.
      b0.p[0] = *(const half8*)(b0Base + kb);
      b0.p[1] = *(const half8*)(b0Base + kb + 8);
      b1.p[0] = *(const half8*)(b1Base + kb);
      b1.p[1] = *(const half8*)(b1Base + kb + 8);

      acc0 = __builtin_amdgcn_wmma_f32_16x16x32_f16(
          false, a.v, false, b0.v, (short)0, acc0, false, false);
      acc1 = __builtin_amdgcn_wmma_f32_16x16x32_f16(
          false, a.v, false, b1.v, (short)0, acc1, false, false);
    }

    // ---- Scatter band elements. D layout: VGPR r -> M = r+8*hi, N = n.
    const int h = h0 + hq;
    float* orow = out + ((size_t)b * CV_K81 + wv * 9 + 4) * CV_HW
                      + (size_t)h * CV_W + w0;
#pragma unroll
    for (int r = 0; r < 8; ++r) {
      const int m = r + 8 * hi;
      // tile0: f2 col = w0-4+n  -> dx = m + 4 - n
      const int dx0 = m + 4 - n;
      if (dx0 >= -4 && dx0 <= 4) orow[(size_t)dx0 * CV_HW + m] = acc0[r];
      // tile1: f2 col = w0+12+n -> dx = m - 12 - n
      const int dx1 = m - 12 - n;
      if (dx1 >= -4 && dx1 <= 4) orow[(size_t)dx1 * CV_HW + m] = acc1[r];
    }
  }
}

extern "C" void kernel_launch(void* const* d_in, const int* in_sizes, int n_in,
                              void* d_out, int out_size, void* d_ws, size_t ws_size,
                              hipStream_t stream) {
  (void)in_sizes; (void)n_in; (void)d_ws; (void)ws_size; (void)out_size;
  const float* f1 = (const float*)d_in[0];
  const float* f2 = (const float*)d_in[1];
  float* out = (float*)d_out;

  dim3 grid(CV_W / 16, CV_H / HPB, CV_B); // (10, 24, 8)
  dim3 block(NTHR);                       // 288 = 9 waves
  cost_volume_wmma_kernel<<<grid, block, SMEM_BYTES, stream>>>(f1, f2, out);
}
